// SpecDetrTransformerEncoderLayer_15307263443644
// MI455X (gfx1250) — matrix operand order repaired and checked by reading, not compiled, not run on hardware
//
#include <hip/hip_runtime.h>
#include <hip/hip_bf16.h>

typedef __attribute__((ext_vector_type(16))) __bf16 v16bf;
typedef __attribute__((ext_vector_type(8)))  float  v8f;

#define BSZ    4
#define NQ     12500
#define DM     256
#define NHEAD  8
#define HD     32
#define DFF    2048
#define NTOK   (BSZ * NQ)      // 50000
#define NCAT   448             // 256 (val) + 128 (off) + 64 (attn)

// ---------------------------------------------------------------------------
// Weight / activation conversion kernels (one-time per launch)
// ---------------------------------------------------------------------------
__global__ void k_cvt_bf16(const float* __restrict__ in, __bf16* __restrict__ out, int n) {
    int i = blockIdx.x * blockDim.x + threadIdx.x;
    if (i < n) out[i] = (__bf16)in[i];
}

// W [K,N] fp32  ->  Wt [N,K] bf16
__global__ void k_transpose_bf16(const float* __restrict__ W, __bf16* __restrict__ Wt, int K, int N) {
    int i = blockIdx.x * blockDim.x + threadIdx.x;
    if (i >= K * N) return;
    int n = i / K, k = i - n * K;
    Wt[i] = (__bf16)W[(long)k * N + n];
}

// concat [w_val | w_off | w_attn] -> Wt [448][256] bf16
__global__ void k_build_wcat(const float* __restrict__ wv, const float* __restrict__ wo,
                             const float* __restrict__ wa, __bf16* __restrict__ Wt) {
    int i = blockIdx.x * blockDim.x + threadIdx.x;
    if (i >= NCAT * DM) return;
    int n = i >> 8, k = i & 255;
    float v;
    if (n < 256)      v = wv[k * 256 + n];
    else if (n < 384) v = wo[k * 128 + (n - 256)];
    else              v = wa[k * 64  + (n - 384)];
    Wt[i] = (__bf16)v;
}

__global__ void k_build_bcat(const float* __restrict__ bv, const float* __restrict__ bo,
                             const float* __restrict__ ba, float* __restrict__ bc) {
    int n = threadIdx.x;
    if (n < 256)      bc[n] = bv[n];
    else if (n < 384) bc[n] = bo[n - 256];
    else if (n < 448) bc[n] = ba[n - 384];
}

// ---------------------------------------------------------------------------
// bf16 WMMA GEMM:  C[M,N] = A[M,K] @ Bt[N,K]^T + bias
// block = 256 threads (8 waves, 4x2), block tile 128x64, wave tile 32x32
// Per K-step per wave: 4 v_wmma per 8 ds_load_b128.
// MODE 0: QKV split epilogue  (C0=value bf16, C1=off f32, C2=aw f32)
// MODE 1: fp32 out + bias     (C0 = float*)
// MODE 2: relu + bf16 out     (C0 = __bf16*)
// ---------------------------------------------------------------------------
union FragAB { v16bf v; uint4 q[2]; };

__device__ __forceinline__ void qkv_store(long m, int n, float v,
                                          __bf16* val, float* off, float* aw) {
    if (n < 256)      val[m * 256 + n]        = (__bf16)v;
    else if (n < 384) off[m * 128 + (n - 256)] = v;
    else              aw [m * 64  + (n - 384)] = v;
}

template <int MODE>
__launch_bounds__(256)
__global__ void k_gemm(const __bf16* __restrict__ A, const __bf16* __restrict__ Bt,
                       const float* __restrict__ bias,
                       void* __restrict__ C0, void* __restrict__ C1, void* __restrict__ C2,
                       int M, int N, int K) {
    constexpr int AP = 56;                 // padded LDS row stride in halves (112B)
    __shared__ __bf16 As[128 * AP];
    __shared__ __bf16 Bs[64 * AP];

    const int t    = threadIdx.x;
    const int lane = t & 31;
    const int wv   = t >> 5;               // wave id 0..7
    const int wm   = wv & 3;               // row sub-tile (32 rows each)
    const int wn   = wv >> 2;              // col sub-tile (32 cols each)
    const int lg   = lane >> 4;            // lane group (0/1)
    const int l15  = lane & 15;

    const int blockM = blockIdx.x * 128;
    const int blockN = blockIdx.y * 64;

    v8f acc00 = {};   // rows +0..15,  cols +0..15
    v8f acc01 = {};   // rows +0..15,  cols +16..31
    v8f acc10 = {};   // rows +16..31, cols +0..15
    v8f acc11 = {};   // rows +16..31, cols +16..31

    // A tile staging: 2 threads per row, 16 halves (32B) each, 128 rows
    const int  arow    = t >> 1;
    const int  ahalf   = t & 1;
    const bool arow_ok = (blockM + arow) < M;
    const long aoff    = (long)(blockM + arow) * K + 16 * ahalf;
    // B tile staging: threads 0..127, 2 threads per n-row, 64 rows
    const int  brow  = (t & 127) >> 1;
    const int  bhalf = t & 1;
    const long boff  = (long)(blockN + brow) * K + 16 * bhalf;

    for (int k0 = 0; k0 < K; k0 += 32) {
        uint4 a0 = make_uint4(0, 0, 0, 0), a1 = a0;
        if (arow_ok) {
            const uint4* src = (const uint4*)(A + aoff + k0);
            a0 = src[0]; a1 = src[1];
            if (k0 + 64 < K)
                __builtin_prefetch((const void*)(A + aoff + k0 + 64), 0, 3);
        }
        *(uint4*)&As[arow * AP + 16 * ahalf]     = a0;
        *(uint4*)&As[arow * AP + 16 * ahalf + 8] = a1;
        if (t < 128) {
            const uint4* src = (const uint4*)(Bt + boff + k0);
            uint4 b0 = src[0], b1 = src[1];
            *(uint4*)&Bs[brow * AP + 16 * bhalf]     = b0;
            *(uint4*)&Bs[brow * AP + 16 * bhalf + 8] = b1;
        }
        __syncthreads();

        // A fragments: lane holds row m=l15, halves k = (h<8 ? h : h+8) + 8*lg
        FragAB a0f, a1f, b0f, b1f;
        const __bf16* ap0 = &As[(wm * 32 + l15) * AP];
        a0f.q[0] = *(const uint4*)(ap0 + 8 * lg);
        a0f.q[1] = *(const uint4*)(ap0 + 16 + 8 * lg);
        const __bf16* ap1 = &As[(wm * 32 + 16 + l15) * AP];
        a1f.q[0] = *(const uint4*)(ap1 + 8 * lg);
        a1f.q[1] = *(const uint4*)(ap1 + 16 + 8 * lg);
        // B fragments: lane holds col n=l15, halves k = (h<8 ? h : h+8) + 8*lg
        const __bf16* bp0 = &Bs[(wn * 32 + l15) * AP + 8 * lg];
        b0f.q[0] = *(const uint4*)(bp0);
        b0f.q[1] = *(const uint4*)(bp0 + 16);
        const __bf16* bp1 = &Bs[(wn * 32 + 16 + l15) * AP + 8 * lg];
        b1f.q[0] = *(const uint4*)(bp1);
        b1f.q[1] = *(const uint4*)(bp1 + 16);

        acc00 = __builtin_amdgcn_wmma_f32_16x16x32_bf16(false, a0f.v, false, b0f.v,
                                                        (short)0, acc00, false, false);
        acc01 = __builtin_amdgcn_wmma_f32_16x16x32_bf16(false, a0f.v, false, b1f.v,
                                                        (short)0, acc01, false, false);
        acc10 = __builtin_amdgcn_wmma_f32_16x16x32_bf16(false, a1f.v, false, b0f.v,
                                                        (short)0, acc10, false, false);
        acc11 = __builtin_amdgcn_wmma_f32_16x16x32_bf16(false, a1f.v, false, b1f.v,
                                                        (short)0, acc11, false, false);
        __syncthreads();
    }

    // Epilogue: C/D layout — VGPR j, lane: m = j + 8*lg, n = lane&15
    const int n0 = blockN + wn * 32 + l15;
    const int n1 = n0 + 16;
    const float bv0 = bias[n0];
    const float bv1 = bias[n1];
    #pragma unroll
    for (int r = 0; r < 2; ++r) {
        const v8f& accA = r ? acc10 : acc00;
        const v8f& accB = r ? acc11 : acc01;
        const int mBase = blockM + wm * 32 + r * 16 + 8 * lg;
        #pragma unroll
        for (int j = 0; j < 8; ++j) {
            const long m = mBase + j;
            if (m >= M) continue;
            const float v0 = accA[j] + bv0;
            const float v1 = accB[j] + bv1;
            if (MODE == 0) {
                qkv_store(m, n0, v0, (__bf16*)C0, (float*)C1, (float*)C2);
                qkv_store(m, n1, v1, (__bf16*)C0, (float*)C1, (float*)C2);
            } else if (MODE == 1) {
                ((float*)C0)[m * N + n0] = v0;
                ((float*)C0)[m * N + n1] = v1;
            } else {
                ((__bf16*)C0)[m * N + n0] = (__bf16)fmaxf(v0, 0.f);
                ((__bf16*)C0)[m * N + n1] = (__bf16)fmaxf(v1, 0.f);
            }
        }
    }
}

// ---------------------------------------------------------------------------
// Deformable attention sampling: one wave per (token, head); lane = channel
// ---------------------------------------------------------------------------
__device__ __forceinline__ float dsample(const __bf16* __restrict__ vb,
                                         int xi, int yi, int H, int W) {
    bool valid = (xi >= 0) & (xi < W) & (yi >= 0) & (yi < H);
    int xc = min(max(xi, 0), W - 1);
    int yc = min(max(yi, 0), H - 1);
    float g = (float)vb[(long)(yc * W + xc) * DM];
    return valid ? g : 0.f;
}

__launch_bounds__(256)
__global__ void k_deform(const __bf16* __restrict__ value, const float* __restrict__ refp,
                         const float* __restrict__ off, const float* __restrict__ awl,
                         __bf16* __restrict__ attn) {
    const int q = blockIdx.x, b = blockIdx.y;
    const int head = threadIdx.x >> 5;
    const int c    = threadIdx.x & 31;
    const long qm  = (long)b * NQ + q;

    // softmax over LEVELS*POINTS = 8 logits of this head
    float lgs[8];
    float mx = -3.0e38f;
    const float* ap = awl + qm * 64 + head * 8;
    #pragma unroll
    for (int i = 0; i < 8; ++i) { lgs[i] = ap[i]; mx = fmaxf(mx, lgs[i]); }
    float s = 0.f;
    #pragma unroll
    for (int i = 0; i < 8; ++i) { lgs[i] = __expf(lgs[i] - mx); s += lgs[i]; }
    const float inv = 1.0f / s;

    const float* op = off  + qm * 128 + head * 16;   // (level,point,xy)
    const float* rp = refp + qm * 4;                 // (level,xy)

    const int Hs[2] = {100, 50}, Ws[2] = {100, 50}, St[2] = {0, 10000};
    float acc = 0.f;
    #pragma unroll
    for (int l = 0; l < 2; ++l) {
        const int H = Hs[l], W = Ws[l];
        const float rx = rp[l * 2 + 0], ry = rp[l * 2 + 1];
        const __bf16* vb = value + ((long)b * NQ + St[l]) * DM + head * HD + c;
        #pragma unroll
        for (int p = 0; p < 4; ++p) {
            const float ox = op[(l * 4 + p) * 2 + 0];
            const float oy = op[(l * 4 + p) * 2 + 1];
            const float x  = (rx + ox / (float)W) * (float)W - 0.5f;
            const float y  = (ry + oy / (float)H) * (float)H - 0.5f;
            const float x0 = floorf(x), y0 = floorf(y);
            const float wx1 = x - x0, wy1 = y - y0;
            const float wx0 = 1.f - wx1, wy0 = 1.f - wy1;
            const int x0i = (int)x0, y0i = (int)y0;
            const float g00 = dsample(vb, x0i,     y0i,     H, W);
            const float g10 = dsample(vb, x0i + 1, y0i,     H, W);
            const float g01 = dsample(vb, x0i,     y0i + 1, H, W);
            const float g11 = dsample(vb, x0i + 1, y0i + 1, H, W);
            acc += (lgs[l * 4 + p] * inv) *
                   (g00 * wx0 * wy0 + g10 * wx1 * wy0 + g01 * wx0 * wy1 + g11 * wx1 * wy1);
        }
    }
    attn[qm * DM + head * HD + c] = (__bf16)acc;
}

// ---------------------------------------------------------------------------
// Residual add + LayerNorm (block = one row of 256)
// ---------------------------------------------------------------------------
template <bool WRITE_BF>
__launch_bounds__(256)
__global__ void k_add_ln(const float* __restrict__ A, const float* __restrict__ B,
                         const float* __restrict__ g, const float* __restrict__ be,
                         float* __restrict__ out_f, __bf16* __restrict__ out_b) {
    __shared__ float r1[256];
    __shared__ float r2[256];
    const long row = blockIdx.x;
    const int  t   = threadIdx.x;
    const float v  = A[row * DM + t] + B[row * DM + t];
    r1[t] = v; r2[t] = v * v;
    __syncthreads();
    for (int s = 128; s > 0; s >>= 1) {
        if (t < s) { r1[t] += r1[t + s]; r2[t] += r2[t + s]; }
        __syncthreads();
    }
    const float mean = r1[0] * (1.f / DM);
    const float var  = r2[0] * (1.f / DM) - mean * mean;
    const float inv  = rsqrtf(var + 1e-5f);
    const float y    = (v - mean) * inv * g[t] + be[t];
    if (out_f) out_f[row * DM + t] = y;
    if (WRITE_BF) out_b[row * DM + t] = (__bf16)y;
}

// ---------------------------------------------------------------------------
// Host launcher
// ---------------------------------------------------------------------------
extern "C" void kernel_launch(void* const* d_in, const int* in_sizes, int n_in,
                              void* d_out, int out_size, void* d_ws, size_t ws_size,
                              hipStream_t stream) {
    const float* query = (const float*)d_in[0];
    const float* refp  = (const float*)d_in[1];
    const float* w_off = (const float*)d_in[2];
    const float* b_off = (const float*)d_in[3];
    const float* w_attn= (const float*)d_in[4];
    const float* b_attn= (const float*)d_in[5];
    const float* w_val = (const float*)d_in[6];
    const float* b_val = (const float*)d_in[7];
    const float* w_out = (const float*)d_in[8];
    const float* b_out = (const float*)d_in[9];
    const float* w1    = (const float*)d_in[10];
    const float* b1    = (const float*)d_in[11];
    const float* w2    = (const float*)d_in[12];
    const float* b2    = (const float*)d_in[13];
    const float* ln0g  = (const float*)d_in[14];
    const float* ln0b  = (const float*)d_in[15];
    const float* ln1g  = (const float*)d_in[16];
    const float* ln1b  = (const float*)d_in[17];
    float* out = (float*)d_out;

    char* ws = (char*)d_ws;
    auto alloc = [&](size_t bytes) -> char* {
        char* p = ws;
        ws += (bytes + 255) & ~(size_t)255;
        return p;
    };
    const int CH = 12800;   // FFN M-chunk (multiple of 16)

    __bf16* q_bf    = (__bf16*)alloc((size_t)NTOK * DM * 2);
    __bf16* val_bf  = (__bf16*)alloc((size_t)NTOK * DM * 2);
    __bf16* attn_bf = (__bf16*)alloc((size_t)NTOK * DM * 2);
    __bf16* x_bf    = (__bf16*)alloc((size_t)NTOK * DM * 2);
    float*  off_f   = (float*) alloc((size_t)NTOK * 128 * 4);
    float*  aw_f    = (float*) alloc((size_t)NTOK * 64 * 4);
    float*  tmp_f   = (float*) alloc((size_t)NTOK * DM * 4);   // attn_out, then FFN2 out
    float*  x_f     = (float*) alloc((size_t)NTOK * DM * 4);
    __bf16* wcat_t  = (__bf16*)alloc((size_t)NCAT * DM * 2);
    __bf16* wout_t  = (__bf16*)alloc((size_t)DM * DM * 2);
    __bf16* w1_t    = (__bf16*)alloc((size_t)DFF * DM * 2);
    __bf16* w2_t    = (__bf16*)alloc((size_t)DM * DFF * 2);
    float*  bcat    = (float*) alloc((size_t)NCAT * 4);
    __bf16* h_bf    = (__bf16*)alloc((size_t)CH * DFF * 2);

    // --- weight prep ---
    k_build_wcat<<<(NCAT * DM + 255) / 256, 256, 0, stream>>>(w_val, w_off, w_attn, wcat_t);
    k_build_bcat<<<1, NCAT, 0, stream>>>(b_val, b_off, b_attn, bcat);
    k_transpose_bf16<<<(DM * DM + 255) / 256, 256, 0, stream>>>(w_out, wout_t, DM, DM);
    k_transpose_bf16<<<(DM * DFF + 255) / 256, 256, 0, stream>>>(w1, w1_t, DM, DFF);
    k_transpose_bf16<<<(DFF * DM + 255) / 256, 256, 0, stream>>>(w2, w2_t, DFF, DM);
    k_cvt_bf16<<<(NTOK * DM + 255) / 256, 256, 0, stream>>>(query, q_bf, NTOK * DM);

    // --- fused value/offset/attn projection: [50000,256] @ [256,448] ---
    k_gemm<0><<<dim3((NTOK + 127) / 128, NCAT / 64), 256, 0, stream>>>(
        q_bf, wcat_t, bcat, val_bf, off_f, aw_f, NTOK, NCAT, DM);

    // --- deformable sampling ---
    k_deform<<<dim3(NQ, BSZ), 256, 0, stream>>>(val_bf, refp, off_f, aw_f, attn_bf);

    // --- output projection + residual LN0 ---
    k_gemm<1><<<dim3((NTOK + 127) / 128, DM / 64), 256, 0, stream>>>(
        attn_bf, wout_t, b_out, tmp_f, nullptr, nullptr, NTOK, DM, DM);
    k_add_ln<true><<<NTOK, 256, 0, stream>>>(query, tmp_f, ln0g, ln0b, x_f, x_bf);

    // --- FFN, chunked over M to bound workspace ---
    for (int m0 = 0; m0 < NTOK; m0 += CH) {
        const int mc = (NTOK - m0 < CH) ? (NTOK - m0) : CH;
        k_gemm<2><<<dim3((mc + 127) / 128, DFF / 64), 256, 0, stream>>>(
            x_bf + (size_t)m0 * DM, w1_t, b1, h_bf, nullptr, nullptr, mc, DFF, DM);
        k_gemm<1><<<dim3((mc + 127) / 128, DM / 64), 256, 0, stream>>>(
            h_bf, w2_t, b2, tmp_f + (size_t)m0 * DM, nullptr, nullptr, mc, DM, DFF);
    }

    // --- residual LN1 -> output ---
    k_add_ln<false><<<NTOK, 256, 0, stream>>>(x_f, tmp_f, ln1g, ln1b, out, nullptr);
}
